// MambaBlockParallel_61744449848169
// MI455X (gfx1250) — matrix operand order repaired and checked by reading, not compiled, not run on hardware
//
#include <hip/hip_runtime.h>

// ---------------------------------------------------------------------------
// Types for CDNA5 WMMA (gfx1250, wave32)
// ---------------------------------------------------------------------------
typedef __attribute__((ext_vector_type(16))) __bf16 v16bf;
typedef __attribute__((ext_vector_type(8)))  __bf16 v8bf;
typedef __attribute__((ext_vector_type(8)))  float  v8f;

#define D_MODEL 1024
#define D_STATE 16
#define BATCH   2
#define SEQ     2048
#define MTOK    (BATCH * SEQ)   // 4096 tokens
#define NCHUNK  32
#define CLEN    (SEQ / NCHUNK)  // 64

// ---------------------------------------------------------------------------
// f32 -> bf16 straight conversion
// ---------------------------------------------------------------------------
__global__ void to_bf16_kernel(const float* __restrict__ in,
                               __bf16* __restrict__ out, int n) {
  int i = blockIdx.x * blockDim.x + threadIdx.x;
  if (i < n) out[i] = (__bf16)in[i];
}

// ---------------------------------------------------------------------------
// W (K x N, f32, row-major) -> Wt (N x K, bf16): WMMA B-fragment becomes a
// contiguous 32B load per lane (lane n holds 16 consecutive K of column n).
// ---------------------------------------------------------------------------
__global__ void transpose_to_bf16_kernel(const float* __restrict__ in,
                                         __bf16* __restrict__ out,
                                         int K, int N) {
  int i = blockIdx.x * blockDim.x + threadIdx.x;
  if (i < K * N) {
    int k = i / N;
    int n = i - k * N;
    out[(size_t)n * K + k] = (__bf16)in[i];
  }
}

// ---------------------------------------------------------------------------
// C(MxN, f32) = A(MxK, bf16) @ Bt(NxK, bf16)^T + bias(N)
// Each wave computes a 16 x (16*TN) strip: one A fragment is reused across
// TN B fragments / accumulators per K-step (register blocking).
// Requires M,N multiples of 16*TN tiling; K multiple of 32. EXEC stays
// all-ones: the bounds guard is wave-uniform.
// ---------------------------------------------------------------------------
template <int TN>
__global__ void gemm_bf16_wmma_kernel(const __bf16* __restrict__ A,
                                      const __bf16* __restrict__ Bt,
                                      const float*  __restrict__ bias,
                                      float* __restrict__ C,
                                      int M, int N, int K) {
  const int lane  = threadIdx.x & 31;
  const int wave  = threadIdx.x >> 5;
  const int gwave = blockIdx.x * (blockDim.x >> 5) + wave;

  const int ngroups = N / (16 * TN);
  const int mtile   = gwave / ngroups;
  const int ngroup  = gwave - mtile * ngroups;
  if (mtile >= (M >> 4)) return;            // wave-uniform

  const int m0 = mtile << 4;
  const int n0 = ngroup * (16 * TN);

  const int r16  = lane & 15;               // M-row (A) / N-col (B)
  const int kgrp = lane >> 4;               // lane-group selects K-subblock

  const __bf16* Arow = A + (size_t)(m0 + r16) * K;   // ISA 16-bit A 16x32
  const __bf16* Brow[TN];
  #pragma unroll
  for (int j = 0; j < TN; ++j)
    Brow[j] = Bt + (size_t)(n0 + j * 16 + r16) * K;  // B 32x16 via Wt

  v8f acc[TN];
  #pragma unroll
  for (int j = 0; j < TN; ++j) acc[j] = (v8f){};

  #pragma unroll 2
  for (int k0 = 0; k0 < K; k0 += 32) {
    // Prefetch next unrolled iteration's streams into cache.
    __builtin_prefetch(Arow + k0 + 64, 0, 1);
    #pragma unroll
    for (int j = 0; j < TN; ++j)
      __builtin_prefetch(Brow[j] + k0 + 64, 0, 1);

    // A fragment: lanes 0-15 hold K=[0..15]; lanes 16-31 hold K=[16..31]
    // (contiguous, matching the B fragment)
    v8bf a_lo = *(const v8bf*)(Arow + k0 + kgrp * 16);
    v8bf a_hi = *(const v8bf*)(Arow + k0 + kgrp * 16 + 8);
    v16bf a = __builtin_shufflevector(a_lo, a_hi,
                                      0, 1, 2, 3, 4, 5, 6, 7,
                                      8, 9, 10, 11, 12, 13, 14, 15);
    #pragma unroll
    for (int j = 0; j < TN; ++j) {
      v16bf b = *(const v16bf*)(Brow[j] + k0 + kgrp * 16);
      acc[j] = __builtin_amdgcn_wmma_f32_16x16x32_bf16(
          /*neg_a=*/false, a, /*neg_b=*/false, b,
          /*c_mod=*/(short)0, acc[j], /*reuse_a=*/false, /*reuse_b=*/false);
    }
  }

  // C/D layout: VGPR r -> M = r (lanes 0-15) / r+8 (lanes 16-31), N = lane&15
  const int rowOff = kgrp * 8;
  const int col    = r16;
  #pragma unroll
  for (int j = 0; j < TN; ++j) {
    const int nc = n0 + j * 16 + col;
    const float bv = bias ? bias[nc] : 0.0f;
    float* Cbase = C + (size_t)(m0 + rowOff) * N + nc;
    #pragma unroll
    for (int r = 0; r < 8; ++r)
      Cbase[(size_t)r * N] = acc[j][r] + bv;
  }
}

// ---------------------------------------------------------------------------
// Chunked parallel scan. Linear recurrence h = a*h + b is associative, so:
//  pass 1: per (b,d,chunk) compute chunk-local state H and decay product P
//  pass 2: per (b,d) 32-step prefix over chunks -> carry-in per chunk
//  pass 3: per (b,d,chunk) replay from carry-in, emit y = (h.C)*sigmoid(z)
// Serial depth drops from 2048 to 64+32; parallelism 65536 threads.
// ---------------------------------------------------------------------------
__device__ __forceinline__ float softplusf(float v) {
  return (v > 20.0f) ? v : log1pf(__expf(v));
}

__global__ void scan_pass1_kernel(const float* __restrict__ xspre,
                                  const float* __restrict__ dtpre,
                                  const float* __restrict__ Bm,
                                  const float* __restrict__ A_log,
                                  float* __restrict__ chunkP,
                                  float* __restrict__ chunkH) {
  const int idx = blockIdx.x * blockDim.x + threadIdx.x; // (b,c,d)
  if (idx >= BATCH * NCHUNK * D_MODEL) return;
  const int d   = idx & (D_MODEL - 1);
  const int tmp = idx >> 10;              // / D_MODEL
  const int c   = tmp & (NCHUNK - 1);
  const int b   = tmp >> 5;               // / NCHUNK

  float Aar[D_STATE], P[D_STATE], h[D_STATE];
  #pragma unroll
  for (int n = 0; n < D_STATE; ++n) {
    Aar[n] = -__expf(A_log[d * D_STATE + n]);
    P[n] = 1.0f;
    h[n] = 0.0f;
  }

  const size_t tok0 = (size_t)b * SEQ + (size_t)c * CLEN;
  for (int l = 0; l < CLEN; ++l) {
    const size_t t = tok0 + l;
    const float dt  = softplusf(dtpre[t * D_MODEL + d]);
    const float xdt = xspre[t * D_MODEL + d] * dt;
    #pragma unroll
    for (int n = 0; n < D_STATE; ++n) {
      const float a = __expf(dt * Aar[n]);
      P[n] *= a;
      h[n] = a * h[n] + xdt * Bm[t * D_STATE + n];
    }
  }

  const size_t base = ((size_t)(b * NCHUNK + c) * D_MODEL + d) * D_STATE;
  #pragma unroll
  for (int n = 0; n < D_STATE; ++n) {
    chunkP[base + n] = P[n];
    chunkH[base + n] = h[n];
  }
}

__global__ void scan_pass2_kernel(const float* __restrict__ chunkP,
                                  const float* __restrict__ chunkH,
                                  float* __restrict__ chunkIn) {
  const int idx = blockIdx.x * blockDim.x + threadIdx.x; // b*D + d
  if (idx >= BATCH * D_MODEL) return;
  const int b = idx >> 10;
  const int d = idx & (D_MODEL - 1);

  float carry[D_STATE];
  #pragma unroll
  for (int n = 0; n < D_STATE; ++n) carry[n] = 0.0f;

  for (int c = 0; c < NCHUNK; ++c) {
    const size_t base = ((size_t)(b * NCHUNK + c) * D_MODEL + d) * D_STATE;
    #pragma unroll
    for (int n = 0; n < D_STATE; ++n) {
      chunkIn[base + n] = carry[n];
      carry[n] = chunkP[base + n] * carry[n] + chunkH[base + n];
    }
  }
}

__global__ void scan_pass3_kernel(const float* __restrict__ zpre,
                                  const float* __restrict__ xspre,
                                  const float* __restrict__ dtpre,
                                  const float* __restrict__ Bm,
                                  const float* __restrict__ Cm,
                                  const float* __restrict__ A_log,
                                  const float* __restrict__ chunkIn,
                                  __bf16* __restrict__ y_bf) {
  const int idx = blockIdx.x * blockDim.x + threadIdx.x; // (b,c,d)
  if (idx >= BATCH * NCHUNK * D_MODEL) return;
  const int d   = idx & (D_MODEL - 1);
  const int tmp = idx >> 10;
  const int c   = tmp & (NCHUNK - 1);
  const int b   = tmp >> 5;

  const size_t base = ((size_t)(b * NCHUNK + c) * D_MODEL + d) * D_STATE;
  float Aar[D_STATE], h[D_STATE];
  #pragma unroll
  for (int n = 0; n < D_STATE; ++n) {
    Aar[n] = -__expf(A_log[d * D_STATE + n]);
    h[n]   = chunkIn[base + n];
  }

  const size_t tok0 = (size_t)b * SEQ + (size_t)c * CLEN;
  for (int l = 0; l < CLEN; ++l) {
    const size_t t = tok0 + l;
    const float z   = 1.0f / (1.0f + __expf(-zpre[t * D_MODEL + d]));
    const float dt  = softplusf(dtpre[t * D_MODEL + d]);
    const float xdt = xspre[t * D_MODEL + d] * dt;
    float y = 0.0f;
    #pragma unroll
    for (int n = 0; n < D_STATE; ++n) {
      h[n] = __expf(dt * Aar[n]) * h[n] + xdt * Bm[t * D_STATE + n];
      y += h[n] * Cm[t * D_STATE + n];
    }
    y_bf[t * D_MODEL + d] = (__bf16)(y * z);
  }
}

// ---------------------------------------------------------------------------
// Host-side orchestration
// ---------------------------------------------------------------------------
static inline size_t alignUp(size_t v, size_t a) { return (v + a - 1) & ~(a - 1); }

extern "C" void kernel_launch(void* const* d_in, const int* in_sizes, int n_in,
                              void* d_out, int out_size, void* d_ws, size_t ws_size,
                              hipStream_t stream) {
  (void)in_sizes; (void)n_in; (void)out_size; (void)ws_size;

  // setup_inputs() order:
  const float* x     = (const float*)d_in[0];
  const float* Wz    = (const float*)d_in[1];
  const float* bz    = (const float*)d_in[2];
  const float* Wx    = (const float*)d_in[3];
  const float* bx    = (const float*)d_in[4];
  const float* WB    = (const float*)d_in[5];
  const float* bB    = (const float*)d_in[6];
  const float* WC    = (const float*)d_in[7];
  const float* bC    = (const float*)d_in[8];
  const float* Wdt   = (const float*)d_in[9];
  const float* bdt   = (const float*)d_in[10];
  const float* A_log = (const float*)d_in[11];
  const float* Wout  = (const float*)d_in[12];
  const float* bout  = (const float*)d_in[13];
  float* out = (float*)d_out;

  const int M = MTOK, D = D_MODEL, N = D_STATE;

  // ---- workspace layout --------------------------------------------------
  char* ws = (char*)d_ws;
  size_t off = 0;
  auto carve = [&](size_t bytes) -> char* {
    char* p = ws + off;
    off = alignUp(off + bytes, 256);
    return p;
  };
  __bf16* x_bf   = (__bf16*)carve((size_t)M * D * 2);
  __bf16* Wz_t   = (__bf16*)carve((size_t)D * D * 2);
  __bf16* Wx_t   = (__bf16*)carve((size_t)D * D * 2);
  __bf16* Wdt_t  = (__bf16*)carve((size_t)D * D * 2);
  __bf16* Wout_t = (__bf16*)carve((size_t)D * D * 2);
  __bf16* WB_t   = (__bf16*)carve((size_t)N * D * 2);
  __bf16* WC_t   = (__bf16*)carve((size_t)N * D * 2);
  float*  zpre   = (float*)carve((size_t)M * D * 4);
  float*  xspre  = (float*)carve((size_t)M * D * 4);
  float*  dtpre  = (float*)carve((size_t)M * D * 4);
  float*  Bmat   = (float*)carve((size_t)M * N * 4);
  float*  Cmat   = (float*)carve((size_t)M * N * 4);
  __bf16* y_bf   = (__bf16*)carve((size_t)M * D * 2);
  float*  chunkP = (float*)carve((size_t)BATCH * NCHUNK * D * N * 4);
  float*  chunkH = (float*)carve((size_t)BATCH * NCHUNK * D * N * 4);
  float*  chunkI = (float*)carve((size_t)BATCH * NCHUNK * D * N * 4);

  // ---- 1) conversions ----------------------------------------------------
  {
    int n = M * D;
    to_bf16_kernel<<<(n + 255) / 256, 256, 0, stream>>>(x, x_bf, n);
  }
  {
    int n = D * D;
    int g = (n + 255) / 256;
    transpose_to_bf16_kernel<<<g, 256, 0, stream>>>(Wz,   Wz_t,   D, D);
    transpose_to_bf16_kernel<<<g, 256, 0, stream>>>(Wx,   Wx_t,   D, D);
    transpose_to_bf16_kernel<<<g, 256, 0, stream>>>(Wdt,  Wdt_t,  D, D);
    transpose_to_bf16_kernel<<<g, 256, 0, stream>>>(Wout, Wout_t, D, D);
    int ns = D * N;
    int gs = (ns + 255) / 256;
    transpose_to_bf16_kernel<<<gs, 256, 0, stream>>>(WB, WB_t, D, N);
    transpose_to_bf16_kernel<<<gs, 256, 0, stream>>>(WC, WC_t, D, N);
  }

  // ---- 2) input projections (WMMA GEMMs) ---------------------------------
  const int threads = 256;                  // 8 waves / block
  const int wavesPerBlock = threads / 32;

  auto launch_gemm4 = [&](const __bf16* A, const __bf16* Bt, const float* bias,
                          float* C, int m, int n, int k) {
    int totalWaves = (m / 16) * (n / 64);
    int blocks = (totalWaves + wavesPerBlock - 1) / wavesPerBlock;
    gemm_bf16_wmma_kernel<4><<<blocks, threads, 0, stream>>>(A, Bt, bias, C, m, n, k);
  };
  auto launch_gemm1 = [&](const __bf16* A, const __bf16* Bt, const float* bias,
                          float* C, int m, int n, int k) {
    int totalWaves = (m / 16) * (n / 16);
    int blocks = (totalWaves + wavesPerBlock - 1) / wavesPerBlock;
    gemm_bf16_wmma_kernel<1><<<blocks, threads, 0, stream>>>(A, Bt, bias, C, m, n, k);
  };

  launch_gemm4(x_bf, Wz_t,  bz,  zpre,  M, D, D);
  launch_gemm4(x_bf, Wx_t,  bx,  xspre, M, D, D);
  launch_gemm4(x_bf, Wdt_t, bdt, dtpre, M, D, D);
  launch_gemm1(x_bf, WB_t,  bB,  Bmat,  M, N, D);
  launch_gemm1(x_bf, WC_t,  bC,  Cmat,  M, N, D);

  // ---- 3) chunked parallel scan ------------------------------------------
  {
    int n1 = BATCH * NCHUNK * D_MODEL;      // 65536 threads
    scan_pass1_kernel<<<(n1 + 255) / 256, 256, 0, stream>>>(
        xspre, dtpre, Bmat, A_log, chunkP, chunkH);
    int n2 = BATCH * D_MODEL;               // 2048 threads
    scan_pass2_kernel<<<(n2 + 255) / 256, 256, 0, stream>>>(
        chunkP, chunkH, chunkI);
    scan_pass3_kernel<<<(n1 + 255) / 256, 256, 0, stream>>>(
        zpre, xspre, dtpre, Bmat, Cmat, A_log, chunkI, y_bf);
  }

  // ---- 4) output projection ----------------------------------------------
  launch_gemm4(y_bf, Wout_t, bout, out, M, D, D);
}
